// MultiHeadSelfAttention_8160437862754
// MI455X (gfx1250) — matrix operand order repaired and checked by reading, not compiled, not run on hardware
//
#include <hip/hip_runtime.h>
#include <hip/hip_bf16.h>
#include <math.h>

typedef __attribute__((ext_vector_type(16))) __bf16 v16bf;
typedef __attribute__((ext_vector_type(8)))  __bf16 v8bf;
typedef __attribute__((ext_vector_type(8)))  float  v8f;
typedef __attribute__((ext_vector_type(4))) unsigned int v4u;
typedef __attribute__((ext_vector_type(8))) int         v8i;
typedef __attribute__((ext_vector_type(4))) int         v4i;

constexpr int Bb = 4, Nn = 2048, Dd = 1024, Hh = 16, DHd = 64;
constexpr int Mtot = Bb * Nn;  // 8192 tokens

// ---------------------------------------------------------------------------
// Feature probes (+ #warning so the compile log reports which paths are live)
// ---------------------------------------------------------------------------
#if __has_builtin(__builtin_amdgcn_global_load_async_to_lds_b128) && \
    __has_builtin(__builtin_amdgcn_s_wait_asynccnt)
#define HAS_ASYNC 1
#warning CDNA5 path report: global_load_async_to_lds ENABLED
#else
#define HAS_ASYNC 0
#warning CDNA5 path report: global_load_async_to_lds DISABLED
#endif

#if __has_builtin(__builtin_amdgcn_tensor_load_to_lds) && \
    __has_builtin(__builtin_amdgcn_s_wait_tensorcnt) && \
    !__has_include(<hip/amd_detail/amd_gfx1250_TDM.h>)
#define HAS_TDM 1
#warning CDNA5 path report: TDM tensor_load_to_lds ENABLED
#else
#define HAS_TDM 0
#warning CDNA5 path report: TDM tensor_load_to_lds DISABLED
#endif

// Async-to-LDS builtin takes (v4i addrspace(1)*, v4i addrspace(3)*, imm, imm).
typedef __attribute__((address_space(1))) v4i* gbl_v4i_p;
typedef __attribute__((address_space(3))) v4i* lds_v4i_p;

static __device__ __forceinline__ void cp16_g2s(void* lds, const void* g) {
#if HAS_ASYNC
  __builtin_amdgcn_global_load_async_to_lds_b128((gbl_v4i_p)g, (lds_v4i_p)lds, 0, 0);
#else
  *(v8bf*)lds = *(const v8bf*)g;
#endif
}
static __device__ __forceinline__ void cp_join() {
#if HAS_ASYNC
  __builtin_amdgcn_s_wait_asynccnt(0);
#endif
}

// ---------------------------------------------------------------------------
// WMMA helpers (CDNA5 v_wmma_f32_16x16x32_bf16, fp32 accumulate)
// ---------------------------------------------------------------------------
static __device__ __forceinline__ v8f wmma_bf16(v16bf a, v16bf b, v8f c) {
  return __builtin_amdgcn_wmma_f32_16x16x32_bf16(false, a, false, b,
                                                 (short)0, c, false, false);
}
static __device__ __forceinline__ v16bf ld_pair(const __bf16* p0, const __bf16* p1) {
  v8bf lo = *(const v8bf*)p0;
  v8bf hi = *(const v8bf*)p1;
  v16bf r;
#pragma unroll
  for (int i = 0; i < 8; ++i) { r[i] = lo[i]; r[i + 8] = hi[i]; }
  return r;
}
static __device__ __forceinline__ v16bf ld_c16(const __bf16* p) {
  return ld_pair(p, p + 16);
}

#if HAS_TDM
// 2D TDM load: tile0 x tile1 elements (2B) with row stride stride0 (elements),
// destination LDS padded via pad_interval/pad_amount (D# group1 fields).
static __device__ __forceinline__ void tdm_load_2d(
    const void* gptr, unsigned lds_byte, unsigned tdim0, unsigned tdim1,
    unsigned tile0, unsigned tile1, unsigned long long stride0,
    unsigned pad_int, unsigned pad_amt) {
  const unsigned long long ga = (unsigned long long)gptr;
  const unsigned long long s1 = stride0 * (unsigned long long)tdim1;
  v4u g0;
  g0[0] = 1u;
  g0[1] = lds_byte;
  g0[2] = (unsigned)ga;
  g0[3] = (unsigned)((ga >> 32) & 0x1FFFFFFull) | (2u << 30);
  v8i g1;
  g1[0] = (int)((1u << 16) | (1u << 20) | (pad_int << 22) | (pad_amt << 25));
  g1[1] = (int)((tdim0 & 0xFFFFu) << 16);
  g1[2] = (int)((tdim0 >> 16) | ((tdim1 & 0xFFFFu) << 16));
  g1[3] = (int)((tdim1 >> 16) | (tile0 << 16));
  g1[4] = (int)(tile1);
  g1[5] = (int)(unsigned)stride0;
  g1[6] = (int)((unsigned)(stride0 >> 32) | (((unsigned)s1 & 0xFFFFu) << 16));
  g1[7] = (int)((unsigned)(s1 >> 16));
  v4i g2; g2[0] = 1; g2[1] = 1; g2[2] = 0; g2[3] = 0;
  v4i g3; g3[0] = 0; g3[1] = 1 << 16; g3[2] = 0; g3[3] = 0;
  __builtin_amdgcn_tensor_load_to_lds(g0, g1, g2, g3, 0);
}
static __device__ __forceinline__ unsigned lds_addr_of(const void* p) {
  return (unsigned)(size_t)p;
}
#endif

// ---------------------------------------------------------------------------
// GEMM 1: out = bf16( X(f32)[8192x1024] @ W(f32)[1024x1024] + bias )
//   mode 0: out[b][h][tok][dh] (Q,K)   mode 1: out[b][h][dh][tok] (V^T)
// Block 128x64, 4 waves x (32x64), K-step 32, double-buffered LDS.
// ---------------------------------------------------------------------------
__global__ __launch_bounds__(128)
void gemm_x_w_bf16(const float* __restrict__ X, const float* __restrict__ W,
                   const float* __restrict__ bias, __bf16* __restrict__ out,
                   int vmode) {
  __shared__ alignas(16) __bf16 As[2][128][40];
  __shared__ alignas(16) __bf16 Bs[2][64][40];
  const int tid = threadIdx.x, lane = tid & 31, wv = tid >> 5;
  const int m0 = blockIdx.x * 128, n0 = blockIdx.y * 64;

  v8f acc[2][4] = {};
  const int arow0 = (wv << 5) + (lane & 15);
  const int koff  = (lane & 16) ? 8 : 0;
  const int kB    = (lane & 16) ? 8 : 0;

  auto stage = [&](int bufi, int k0) {
    const float* src = X + (size_t)(m0 + tid) * Dd + k0;
#pragma unroll
    for (int i = 0; i < 32; ++i) As[bufi][tid][i] = (__bf16)src[i];
    const int kr = tid >> 2, nn = (tid & 3) << 4;
    const float* wsrc = W + (size_t)(k0 + kr) * Dd + n0 + nn;
#pragma unroll
    for (int i = 0; i < 16; ++i) Bs[bufi][nn + i][kr] = (__bf16)wsrc[i];
  };

  stage(0, 0);
  constexpr int KT = Dd / 32;
  for (int kt = 0; kt < KT; ++kt) {
    __syncthreads();
    const int cur = kt & 1;
    if (kt + 1 < KT) stage(cur ^ 1, (kt + 1) * 32);  // overlaps with WMMAs

    v16bf a0 = ld_pair(&As[cur][arow0][koff], &As[cur][arow0][koff + 16]);
    v16bf a1 = ld_pair(&As[cur][arow0 + 16][koff], &As[cur][arow0 + 16][koff + 16]);
    v16bf bf[4];
#pragma unroll
    for (int c = 0; c < 4; ++c)
      bf[c] = ld_c16(&Bs[cur][(c << 4) + (lane & 15)][kB]);
#pragma unroll
    for (int c = 0; c < 4; ++c) {
      acc[0][c] = wmma_bf16(a0, bf[c], acc[0][c]);
      acc[1][c] = wmma_bf16(a1, bf[c], acc[1][c]);
    }
  }

#pragma unroll
  for (int rb = 0; rb < 2; ++rb) {
#pragma unroll
    for (int c = 0; c < 4; ++c) {
#pragma unroll
      for (int r = 0; r < 8; ++r) {
        const int mrow = (rb << 4) + ((lane >> 4) << 3) + r;
        const int grow = m0 + (wv << 5) + mrow;
        const int n = n0 + (c << 4) + (lane & 15);
        const float v = acc[rb][c][r] + bias[n];
        const int bidx = grow >> 11, tok = grow & (Nn - 1);
        const int h = n >> 6, dh = n & 63;
        size_t idx;
        if (vmode) idx = (((size_t)bidx * Hh + h) * DHd + dh) * Nn + tok;
        else       idx = (((size_t)bidx * Hh + h) * Nn + tok) * DHd + dh;
        out[idx] = (__bf16)v;
      }
    }
  }
}

// ---------------------------------------------------------------------------
// Fused causal flash attention. 1 block = one (b,h) x 64 queries, 4 waves.
// K/V tiles double-buffered; staged via TDM / async-to-LDS / plain copies.
// ---------------------------------------------------------------------------
__global__ __launch_bounds__(128)
void attn_fused(const __bf16* __restrict__ Q, const __bf16* __restrict__ K,
                const __bf16* __restrict__ Vt, __bf16* __restrict__ Oout) {
  __shared__ alignas(16) __bf16 Ks[2][32][72];   // [key][dh], 128B + 16B pad
  __shared__ alignas(16) __bf16 Vs[2][64][40];   // [dh][key],  64B + 16B pad
  __shared__ alignas(16) float  Ss[4][16][36];
  __shared__ alignas(16) __bf16 Ps[4][16][40];
  __shared__ alignas(16) float  Al[4][16];

  const int tid = threadIdx.x, lane = tid & 31, wv = tid >> 5;
  const int bh  = blockIdx.x >> 5;
  const int q0b = (blockIdx.x & 31) * 64;
  const int q0  = q0b + (wv << 4);
  const __bf16* Qh = Q  + (size_t)bh * Nn * DHd;
  const __bf16* Kh = K  + (size_t)bh * Nn * DHd;
  const __bf16* Vh = Vt + (size_t)bh * DHd * Nn;

  const int qrow = lane & 15;
  const int koff = (lane & 16) ? 8 : 0;
  const int kB   = (lane & 16) ? 8 : 0;

  const __bf16* qp = Qh + (size_t)(q0 + qrow) * DHd;
  const v16bf qa0 = ld_pair(qp + koff,      qp + koff + 16);
  const v16bf qa1 = ld_pair(qp + 32 + koff, qp + 32 + koff + 16);

  v8f o[4] = {};
  float m_i = -1e30f, l_i = 0.f;

  auto stageKV = [&](int bufi, int kb) {
#if HAS_TDM
    if (wv == 0) {  // one wave drives the Tensor Data Mover for both tiles
      tdm_load_2d(Kh + (size_t)kb * DHd, lds_addr_of(&Ks[bufi][0][0]),
                  DHd, Nn, DHd, 32, DHd, /*pad_int=*/4, /*pad_amt=*/3);
      tdm_load_2d(Vh + kb, lds_addr_of(&Vs[bufi][0][0]),
                  Nn, DHd, 32, DHd, Nn, /*pad_int=*/3, /*pad_amt=*/3);
    }
#else
    const int kr = tid >> 2, dh0 = (tid & 3) << 4;
    const __bf16* ks = Kh + (size_t)(kb + kr) * DHd + dh0;
    cp16_g2s(&Ks[bufi][kr][dh0], ks);
    cp16_g2s(&Ks[bufi][kr][dh0 + 8], ks + 8);
    const int vr = tid >> 1, c0 = (tid & 1) << 4;
    const __bf16* vs = Vh + (size_t)vr * Nn + kb + c0;
    cp16_g2s(&Vs[bufi][vr][c0], vs);
    cp16_g2s(&Vs[bufi][vr][c0 + 8], vs + 8);
#endif
  };
  auto joinKV = [&]() {
#if HAS_TDM
    if (wv == 0) __builtin_amdgcn_s_wait_tensorcnt(0);
#else
    cp_join();
#endif
  };

  const int ntiles = (q0b >> 5) + 2;
  stageKV(0, 0);
  for (int t = 0; t < ntiles; ++t) {
    joinKV();
    __syncthreads();
    const int cur = t & 1;
    if (t + 1 < ntiles) stageKV(cur ^ 1, (t + 1) << 5);  // DMA overlaps compute

    // Scores: S = (Q K^T) * 1/sqrt(64), causal-masked, into per-wave LDS
#pragma unroll
    for (int c = 0; c < 2; ++c) {
      v16bf b0 = ld_c16(&Ks[cur][(c << 4) + (lane & 15)][kB]);
      v16bf b1 = ld_c16(&Ks[cur][(c << 4) + (lane & 15)][32 + kB]);
      v8f s = {};
      s = wmma_bf16(qa0, b0, s);
      s = wmma_bf16(qa1, b1, s);
      const int kb = t << 5;
#pragma unroll
      for (int r = 0; r < 8; ++r) {
        const int mr  = ((lane >> 4) << 3) + r;
        const int key = kb + (c << 4) + (lane & 15);
        Ss[wv][mr][(c << 4) + (lane & 15)] =
            (key <= q0 + mr) ? s[r] * 0.125f : -1e30f;
      }
    }
    __syncthreads();

    // Online softmax: lane m (<16) owns query row m of its wave
    if (lane < 16) {
      const float* srow = &Ss[wv][lane][0];
      float mx = -1e30f;
#pragma unroll
      for (int j = 0; j < 32; ++j) mx = fmaxf(mx, srow[j]);
      const float mnew = fmaxf(m_i, mx);
      const float corr = __expf(m_i - mnew);
      float sum = 0.f;
#pragma unroll
      for (int j = 0; j < 32; ++j) {
        const float p = __expf(srow[j] - mnew);
        sum += p;
        Ps[wv][lane][j] = (__bf16)p;
      }
      l_i = l_i * corr + sum;
      m_i = mnew;
      Al[wv][lane] = corr;
    }
    __syncthreads();

    // Rescale accumulators, then O += P @ V (V fragments preloaded)
    const v16bf pa = ld_pair(&Ps[wv][qrow][koff], &Ps[wv][qrow][koff + 16]);
    v16bf vbf[4];
#pragma unroll
    for (int c = 0; c < 4; ++c)
      vbf[c] = ld_c16(&Vs[cur][(c << 4) + (lane & 15)][kB]);
#pragma unroll
    for (int r = 0; r < 8; ++r) {
      const float al = Al[wv][((lane >> 4) << 3) + r];
#pragma unroll
      for (int c = 0; c < 4; ++c) o[c][r] = o[c][r] * al;
    }
#pragma unroll
    for (int c = 0; c < 4; ++c) o[c] = wmma_bf16(pa, vbf[c], o[c]);
  }

  __syncthreads();
  if (lane < 16) Al[wv][lane] = 1.f / l_i;
  __syncthreads();

  const int b = bh >> 4, h = bh & 15;
#pragma unroll
  for (int r = 0; r < 8; ++r) {
    const float inv = Al[wv][((lane >> 4) << 3) + r];
    const int tok = q0 + ((lane >> 4) << 3) + r;
#pragma unroll
    for (int c = 0; c < 4; ++c) {
      const int dh = (c << 4) + (lane & 15);
      Oout[((size_t)b * Nn + tok) * Dd + h * DHd + dh] = (__bf16)(o[c][r] * inv);
    }
  }
}

// ---------------------------------------------------------------------------
// GEMM 2: out(f32) = A(bf16)[8192x1024] @ Wo(f32)[1024x1024] + bo
// Double-buffered; A tile staged with async global->LDS copies.
// ---------------------------------------------------------------------------
__global__ __launch_bounds__(128)
void gemm_a_wo_f32(const __bf16* __restrict__ A, const float* __restrict__ W,
                   const float* __restrict__ bias, float* __restrict__ out) {
  __shared__ alignas(16) __bf16 As[2][128][40];
  __shared__ alignas(16) __bf16 Bs[2][64][40];
  const int tid = threadIdx.x, lane = tid & 31, wv = tid >> 5;
  const int m0 = blockIdx.x * 128, n0 = blockIdx.y * 64;

  v8f acc[2][4] = {};
  const int arow0 = (wv << 5) + (lane & 15);
  const int koff  = (lane & 16) ? 8 : 0;
  const int kB    = (lane & 16) ? 8 : 0;

  auto stage = [&](int bufi, int k0) {
    const __bf16* src = A + (size_t)(m0 + tid) * Dd + k0;
    cp16_g2s(&As[bufi][tid][0],  src);
    cp16_g2s(&As[bufi][tid][8],  src + 8);
    cp16_g2s(&As[bufi][tid][16], src + 16);
    cp16_g2s(&As[bufi][tid][24], src + 24);
    const int kr = tid >> 2, nn = (tid & 3) << 4;
    const float* wsrc = W + (size_t)(k0 + kr) * Dd + n0 + nn;
#pragma unroll
    for (int i = 0; i < 16; ++i) Bs[bufi][nn + i][kr] = (__bf16)wsrc[i];
  };

  stage(0, 0);
  constexpr int KT = Dd / 32;
  for (int kt = 0; kt < KT; ++kt) {
    cp_join();
    __syncthreads();
    const int cur = kt & 1;
    if (kt + 1 < KT) stage(cur ^ 1, (kt + 1) * 32);  // async DMA overlaps WMMAs

    v16bf a0 = ld_pair(&As[cur][arow0][koff], &As[cur][arow0][koff + 16]);
    v16bf a1 = ld_pair(&As[cur][arow0 + 16][koff], &As[cur][arow0 + 16][koff + 16]);
    v16bf bf[4];
#pragma unroll
    for (int c = 0; c < 4; ++c)
      bf[c] = ld_c16(&Bs[cur][(c << 4) + (lane & 15)][kB]);
#pragma unroll
    for (int c = 0; c < 4; ++c) {
      acc[0][c] = wmma_bf16(a0, bf[c], acc[0][c]);
      acc[1][c] = wmma_bf16(a1, bf[c], acc[1][c]);
    }
  }

#pragma unroll
  for (int rb = 0; rb < 2; ++rb) {
#pragma unroll
    for (int c = 0; c < 4; ++c) {
#pragma unroll
      for (int r = 0; r < 8; ++r) {
        const int mrow = (rb << 4) + ((lane >> 4) << 3) + r;
        const int grow = m0 + (wv << 5) + mrow;
        const int n = n0 + (c << 4) + (lane & 15);
        out[(size_t)grow * Dd + n] = acc[rb][c][r] + bias[n];
      }
    }
  }
}

// ---------------------------------------------------------------------------
extern "C" void kernel_launch(void* const* d_in, const int* in_sizes, int n_in,
                              void* d_out, int out_size, void* d_ws, size_t ws_size,
                              hipStream_t stream) {
  const float* x  = (const float*)d_in[0];
  const float* Wq = (const float*)d_in[1];
  const float* bq = (const float*)d_in[2];
  const float* Wk = (const float*)d_in[3];
  const float* bk = (const float*)d_in[4];
  const float* Wv = (const float*)d_in[5];
  const float* bv = (const float*)d_in[6];
  const float* Wo = (const float*)d_in[7];
  const float* bo = (const float*)d_in[8];
  float* out = (float*)d_out;

  const size_t elems = (size_t)Mtot * Dd;
  __bf16* q_ws = (__bf16*)d_ws;
  __bf16* k_ws = q_ws + elems;
  __bf16* v_ws = k_ws + elems;
  __bf16* a_ws = v_ws + elems;

  dim3 gg(Mtot / 128, Dd / 64);
  gemm_x_w_bf16<<<gg, 128, 0, stream>>>(x, Wq, bq, q_ws, 0);
  gemm_x_w_bf16<<<gg, 128, 0, stream>>>(x, Wk, bk, k_ws, 0);
  gemm_x_w_bf16<<<gg, 128, 0, stream>>>(x, Wv, bv, v_ws, 1);
  attn_fused<<<Bb * Hh * (Nn / 64), 128, 0, stream>>>(q_ws, k_ws, v_ws, a_ws);
  gemm_a_wo_f32<<<gg, 128, 0, stream>>>(a_ws, Wo, bo, out);
}
